// LSTM_61821759258810
// MI455X (gfx1250) — matrix operand order repaired and checked by reading, not compiled, not run on hardware
//
#include <hip/hip_runtime.h>

typedef __attribute__((ext_vector_type(16))) _Float16 v16h;
typedef __attribute__((ext_vector_type(8)))  _Float16 v8h;
typedef __attribute__((ext_vector_type(8)))  float    v8f;
typedef __attribute__((ext_vector_type(2)))  float    v2f;

#define T_LEN 65536
#define IN_DIM 5
#define H_DIM 100
#define G_DIM 400   // 4*H
#define KPAD 128    // H padded to multiple of 32 for WMMA K

// Fast device transcendentals on native trans units (v_exp_f32 / v_rcp_f32).
__device__ __forceinline__ float fast_sigmoid(float x) {
    const float e = __builtin_amdgcn_exp2f(-1.4426950408889634f * x);
    return __builtin_amdgcn_rcpf(1.0f + e);
}
__device__ __forceinline__ float fast_tanh(float x) {
    const float e = __builtin_amdgcn_exp2f(-2.8853900817779268f * x);
    return __builtin_amdgcn_rcpf(1.0f + e) * 2.0f - 1.0f;
}

// ---------------------------------------------------------------------------
// Phase A: xg0[t][g] = sum_i x[t][i]*Wih0[g][i] + bih0[g] + bhh0[g]   (K=5)
// ---------------------------------------------------------------------------
__global__ void xg0_kernel(const float* __restrict__ x,
                           const float* __restrict__ Wih0,
                           const float* __restrict__ bih,
                           const float* __restrict__ bhh,
                           float* __restrict__ xg) {
    int idx = blockIdx.x * blockDim.x + threadIdx.x;
    if (idx >= T_LEN * G_DIM) return;
    int t = idx / G_DIM;
    int g = idx - t * G_DIM;
    const float* xp = x + t * IN_DIM;
    const float* wp = Wih0 + g * IN_DIM;
    float acc = bih[g] + bhh[g];
#pragma unroll
    for (int i = 0; i < IN_DIM; ++i) acc += xp[i] * wp[i];
    xg[idx] = acc;
}

// ---------------------------------------------------------------------------
// Convert Wih1 (400x100 f32) -> f16, K zero-padded to 128 (row-major [g][k])
// ---------------------------------------------------------------------------
__global__ void cvt_wih1_kernel(const float* __restrict__ Wih1,
                                _Float16* __restrict__ Wf16) {
    int idx = blockIdx.x * blockDim.x + threadIdx.x;
    if (idx >= G_DIM * KPAD) return;
    int g = idx >> 7;
    int k = idx & (KPAD - 1);
    Wf16[idx] = (k < H_DIM) ? (_Float16)Wih1[g * H_DIM + k] : (_Float16)0.0f;
}

// ---------------------------------------------------------------------------
// Serial LSTM recurrence: one workgroup (416 thr = 13 wave32).
// Lanes 0..399: gate g = tid; Whh row register-resident as 50 float2 pairs;
// h broadcast via LDS (conflict-free); dot product runs on v_pk_fma_f32
// (2 fp32 FMA / instruction) over two independent accumulator chains.
// MODE 0: store h as f16 into KPAD-padded rows (feeds WMMA GEMM).
// MODE 1: store h as f32 (feeds final linear).
// ---------------------------------------------------------------------------
template <int MODE>
__global__ __launch_bounds__(416, 1)
void lstm_recur_kernel(const float* __restrict__ xg,
                       const float* __restrict__ Whh,
                       _Float16* __restrict__ hf16,
                       float* __restrict__ hf32) {
    __shared__ __align__(16) float h_s[104];
    __shared__ float gates_s[G_DIM];

    const int g = threadIdx.x;

    // Register-resident Whh row for this gate, as packed float2 pairs.
    v2f w2[H_DIM / 2];
    if (g < G_DIM) {
        const v2f* wp = (const v2f*)(Whh + g * H_DIM);
#pragma unroll
        for (int k = 0; k < H_DIM / 2; ++k) w2[k] = wp[k];
    }
    if (g < 104) h_s[g] = 0.0f;

    float c_state = 0.0f;
    float xg_cur = (g < G_DIM) ? xg[g] : 0.0f;
    __syncthreads();

    for (int t = 0; t < T_LEN; ++t) {
        float xg_nxt = 0.0f;
        if (g < G_DIM) {
            // Software pipeline: issue next step's gate-bias load now, hide
            // its latency under the dot product below.
            const int tn = (t + 1 < T_LEN) ? (t + 1) : t;
            xg_nxt = xg[(size_t)tn * G_DIM + g];
            const int tp = (t + 8 < T_LEN) ? (t + 8) : t;
            __builtin_prefetch(xg + (size_t)tp * G_DIM + g, 0, 1);

            // Two independent pk-FMA chains: 50 v_pk_fma_f32 = 100 fp32 MACs.
            v2f acc0 = {0.0f, 0.0f};
            v2f acc1 = {0.0f, 0.0f};
            const v2f* h2p = (const v2f*)h_s;
#pragma unroll
            for (int k = 0; k < H_DIM; k += 4) {
                const v2f hlo = h2p[k / 2];        // ds_load pairs; all lanes
                const v2f hhi = h2p[k / 2 + 1];    // same addr -> broadcast
                asm("v_pk_fma_f32 %0, %1, %2, %0"
                    : "+v"(acc0) : "v"(w2[k / 2]), "v"(hlo));
                asm("v_pk_fma_f32 %0, %1, %2, %0"
                    : "+v"(acc1) : "v"(w2[k / 2 + 1]), "v"(hhi));
            }
            const float acc = xg_cur + ((acc0.x + acc0.y) + (acc1.x + acc1.y));

            // PyTorch gate order i,f,g,o : gate group 2 is tanh, rest sigmoid.
            float act;
            if (g >= 2 * H_DIM && g < 3 * H_DIM) act = fast_tanh(acc);
            else                                 act = fast_sigmoid(acc);
            gates_s[g] = act;
        }
        __syncthreads();
        if (g < H_DIM) {
            const float iv = gates_s[g];
            const float fv = gates_s[H_DIM + g];
            const float gv = gates_s[2 * H_DIM + g];
            const float ov = gates_s[3 * H_DIM + g];
            c_state = fv * c_state + iv * gv;
            const float hv = ov * fast_tanh(c_state);
            h_s[g] = hv;
            if (MODE == 0) hf16[(size_t)t * KPAD + g] = (_Float16)hv;
            else           hf32[(size_t)t * H_DIM + g] = hv;
        } else if (MODE == 0 && g < KPAD) {
            hf16[(size_t)t * KPAD + g] = (_Float16)0.0f;   // K zero-pad
        }
        __syncthreads();
        xg_cur = xg_nxt;
    }
}

// ---------------------------------------------------------------------------
// Phase C: xg1 = h1f16 (T x 128) @ Wih1f16^T (128 x 400) + bih1 + bhh1
// One 16x16 C tile per wave via v_wmma_f32_16x16x32_f16, K = 4 x 32.
// A per-lane layout used here (16-bit A 16x32, K contiguous): lane<16 holds K 0..15
// of row M=lane; lane>=16 holds K 16..31 of row M=lane-16.
// B per-lane layout (32x16): lane<16 holds K 0..15 of col N=lane; lane>=16
// holds K 16..31 of col N=lane-16.
// ---------------------------------------------------------------------------
__global__ __launch_bounds__(256)
void xg1_wmma_kernel(const _Float16* __restrict__ A,   // T x KPAD
                     const _Float16* __restrict__ Bt,  // G_DIM x KPAD (Wih1 rows)
                     const float* __restrict__ bih,
                     const float* __restrict__ bhh,
                     float* __restrict__ xg) {         // T x G_DIM
    const int lane = threadIdx.x & 31;
    const int wave = threadIdx.x >> 5;
    const int wid  = blockIdx.x * (blockDim.x >> 5) + wave;
    const int tileN = wid % (G_DIM / 16);              // 25 tiles
    const int tileM = wid / (G_DIM / 16);              // 4096 tiles
    const int half16 = lane >> 4;                      // 0 | 1
    const int l16    = lane & 15;
    const int rowA = tileM * 16 + l16;
    const int colB = tileN * 16 + l16;

    v8f c = {};
#pragma unroll
    for (int kc = 0; kc < KPAD / 32; ++kc) {
        const int kbase = kc * 32;
        // A fragment: two contiguous 8-half (16B) groups.
        const _Float16* ap = A + (size_t)rowA * KPAD + kbase + half16 * 16;
        const v8h a_lo = *(const v8h*)(ap);
        const v8h a_hi = *(const v8h*)(ap + 8);
        v16h a;
#pragma unroll
        for (int i = 0; i < 8; ++i) { a[i] = a_lo[i]; a[8 + i] = a_hi[i]; }
        // B fragment: 16 contiguous halfs of Wih1 row (= column of B).
        const _Float16* bp = Bt + (size_t)colB * KPAD + kbase + half16 * 16;
        const v16h b = *(const v16h*)(bp);
        c = __builtin_amdgcn_wmma_f32_16x16x32_f16(
                false, a, false, b, (short)0, c, false, false);
    }
    // C layout: VGPR r, lane -> M = r + half16*8, N = lane&15.
    const float bias = bih[colB] + bhh[colB];
#pragma unroll
    for (int r = 0; r < 8; ++r) {
        const int m = tileM * 16 + r + half16 * 8;
        xg[(size_t)m * G_DIM + colB] = c[r] + bias;
    }
}

// ---------------------------------------------------------------------------
// Phase E: out[t] = sigmoid(h2[t] . Wlin + blin)
// ---------------------------------------------------------------------------
__global__ void final_kernel(const float* __restrict__ h2,
                             const float* __restrict__ Wlin,
                             const float* __restrict__ blin,
                             float* __restrict__ out) {
    int t = blockIdx.x * blockDim.x + threadIdx.x;
    if (t >= T_LEN) return;
    const float* hp = h2 + (size_t)t * H_DIM;
    float acc = blin[0];
#pragma unroll
    for (int k = 0; k < H_DIM; ++k) acc += hp[k] * Wlin[k];
    out[t] = fast_sigmoid(acc);
}

// ---------------------------------------------------------------------------
extern "C" void kernel_launch(void* const* d_in, const int* in_sizes, int n_in,
                              void* d_out, int out_size, void* d_ws, size_t ws_size,
                              hipStream_t stream) {
    const float* x    = (const float*)d_in[0];
    const float* Wih0 = (const float*)d_in[1];
    const float* Whh0 = (const float*)d_in[2];
    const float* bih0 = (const float*)d_in[3];
    const float* bhh0 = (const float*)d_in[4];
    const float* Wih1 = (const float*)d_in[5];
    const float* Whh1 = (const float*)d_in[6];
    const float* bih1 = (const float*)d_in[7];
    const float* bhh1 = (const float*)d_in[8];
    const float* Wlin = (const float*)d_in[9];
    const float* blin = (const float*)d_in[10];
    float* out = (float*)d_out;

    char* ws = (char*)d_ws;
    const size_t xg_bytes  = (size_t)T_LEN * G_DIM * sizeof(float);   // 104.9 MB (reused for both layers)
    const size_t h1_bytes  = (size_t)T_LEN * KPAD * sizeof(_Float16); // 16.8 MB
    const size_t h2_bytes  = (size_t)T_LEN * H_DIM * sizeof(float);   // 26.2 MB
    float*    xg  = (float*)ws;
    _Float16* h1  = (_Float16*)(ws + xg_bytes);
    float*    h2  = (float*)(ws + xg_bytes + h1_bytes);
    _Float16* W1f = (_Float16*)(ws + xg_bytes + h1_bytes + h2_bytes);

    // A: input projection for layer 0 (biases folded in)
    xg0_kernel<<<(T_LEN * G_DIM + 255) / 256, 256, 0, stream>>>(x, Wih0, bih0, bhh0, xg);
    // prep: Wih1 -> f16 padded
    cvt_wih1_kernel<<<(G_DIM * KPAD + 255) / 256, 256, 0, stream>>>(Wih1, W1f);
    // B: layer-0 serial recurrence -> h1 (f16, K-padded)
    lstm_recur_kernel<0><<<1, 416, 0, stream>>>(xg, Whh0, h1, nullptr);
    // C: layer-1 input projection via WMMA (overwrites xg)
    {
        const int waves = (T_LEN / 16) * (G_DIM / 16);   // 102400
        xg1_wmma_kernel<<<waves / 8, 256, 0, stream>>>(h1, W1f, bih1, bhh1, xg);
    }
    // D: layer-1 serial recurrence -> h2 (f32)
    lstm_recur_kernel<1><<<1, 416, 0, stream>>>(xg, Whh1, nullptr, h2);
    // E: final linear + sigmoid
    final_kernel<<<(T_LEN + 255) / 256, 256, 0, stream>>>(h2, Wlin, blin, out);
}
